// VRNN_41772851921255
// MI455X (gfx1250) — matrix operand order repaired and checked by reading, not compiled, not run on hardware
//
#include <hip/hip_runtime.h>
#include <hip/hip_bf16.h>
#include <math.h>

typedef unsigned short u16;
typedef unsigned int   u32;
typedef unsigned long long u64;

typedef __attribute__((ext_vector_type(16))) __bf16 v16bf;
typedef __attribute__((ext_vector_type(8)))  float  v8f;
typedef __attribute__((ext_vector_type(4)))  unsigned int v4u;
typedef __attribute__((ext_vector_type(8)))  int v8i;
typedef __attribute__((ext_vector_type(4)))  int v4i;

// ---------------- helpers ----------------
__device__ __forceinline__ u16 f2bf(float f) {
  u32 u = __float_as_uint(f);
  u32 r = u + 0x7FFFu + ((u >> 16) & 1u);   // round-to-nearest-even
  return (u16)(r >> 16);
}

struct Frag { union { v16bf v; uint4 u[2]; }; };

__device__ __forceinline__ v16bf load_frag16(const u16* p0, const u16* p1) {
  Frag f;
  f.u[0] = *reinterpret_cast<const uint4*>(p0);
  f.u[1] = *reinterpret_cast<const uint4*>(p1);
  return f.v;
}

__device__ __forceinline__ float sigmoidf_(float x) { return 1.f / (1.f + expf(-x)); }
__device__ __forceinline__ float softplusf_(float x) { return (x > 20.f) ? x : log1pf(expf(x)); }

// ---------------- Tensor Data Mover: B-tile DMA ----------------
// DMA a 2D tile of 64 rows (N) x 32 elems (K) of bf16 from row-major W
// (row stride = ldw elements) into LDS at byte offset ldsByte.
// D# layout per CDNA5 ISA ch.8 (group0 128b, group1 256b, groups 2/3 zero = 2D).
// This toolchain exposes the 6-arg builtin:
//   (uint32x4 g0, int32x8 g1, int32x4, int32x4, int32x8, i32 cpol)
__device__ __forceinline__ void tdm_load_b_tile(u32 ldsByte, const u16* gsrc,
                                                u32 tdim0, u32 tdim1, u32 stride0) {
  u64 ga = (u64)(size_t)gsrc;
  v4u g0;
  g0.x = 1u;                                              // count=1 (valid descriptor)
  g0.y = ldsByte;                                         // lds_addr [63:32]
  g0.z = (u32)ga;                                         // global_addr [95:64]
  g0.w = ((u32)(ga >> 32) & 0x01FFFFFFu) | 0x80000000u;   // addr[56:32] | type=2<<30
  v8i g1;
  g1[0] = (int)(1u << 16);                                // data_size=1 (2 bytes)
  g1[1] = (int)((tdim0 & 0xFFFFu) << 16);                 // tensor_dim0[15:0] @ bit48
  g1[2] = (int)((tdim0 >> 16) | ((tdim1 & 0xFFFFu) << 16)); // tdim0 hi | tdim1 lo
  g1[3] = (int)((tdim1 >> 16) | (32u << 16));             // tdim1 hi | tile_dim0=32
  g1[4] = 64;                                             // tile_dim1=64, tile_dim2=0
  g1[5] = (int)stride0;                                   // tensor_dim0_stride[31:0]
  g1[6] = 0;                                              // stride0 hi=0 | stride1 lo=0
  g1[7] = 0;
  v4i gz4 = {0, 0, 0, 0};
  v8i gz8 = {0, 0, 0, 0, 0, 0, 0, 0};
  __builtin_amdgcn_tensor_load_to_lds(g0, g1, gz4, gz4, gz8, 0);
}

// ---------------- WMMA GEMM (TDM-staged B) ----------------
// Y[512, N] = act( A[512, K](bf16) @ W[N, K](bf16)^T + bias )
// Wave: 16(M) x 64(N); 8 waves along M -> block tile 128x64.
// B tile (64N x 32K) double-buffered in LDS via tensor_load_to_lds issued by wave 0.
// act: 0 linear(f32), 1 relu(bf16), 2 sigmoid(f32), 3 softplus(f32), 4 softplus+1(f32)
__global__ __launch_bounds__(256) void gemm_wmma_bf16(
    const u16* __restrict__ A, int lda,
    const u16* __restrict__ W, int ldw, int Wn,
    const float* __restrict__ bias, int nbias,
    float* __restrict__ outF, u16* __restrict__ outB, int ldo,
    int K, int act)
{
  __shared__ u16 bsm[2][64 * 32];

  const int lane = threadIdx.x & 31;
  const int wv   = threadIdx.x >> 5;
  const int hl   = lane >> 4;
  const int l16  = lane & 15;
  const int mBase = blockIdx.x * 128 + wv * 16;
  const int nBase = blockIdx.y * 64;

  const u32 ldsBase = (u32)(size_t)(&bsm[0][0]);   // low 32 bits of flat = LDS byte addr
  const u16* wTile  = W + (size_t)nBase * ldw;     // tile start (k advances per stage)
  const int  S      = K >> 5;                      // number of 32-wide K stages

  // A frag: lane holds row mBase+l16; chunks {0..7,16..23} (hl=0) / {8..15,24..31}
  const u16* aP = A + (size_t)(mBase + l16) * lda + (hl << 3);

  v8f acc0 = {}, acc1 = {}, acc2 = {}, acc3 = {};

  if (wv == 0) tdm_load_b_tile(ldsBase, wTile, (u32)ldw, (u32)Wn, (u32)ldw);

  for (int s = 0; s < S; ++s) {
    if (wv == 0) {
      if (s + 1 < S) {
        tdm_load_b_tile(ldsBase + (u32)(((s + 1) & 1) * (64 * 32 * 2)),
                        wTile + (size_t)(s + 1) * 32, (u32)ldw, (u32)Wn, (u32)ldw);
        __builtin_amdgcn_s_wait_tensorcnt((short)1);   // stage s landed (in-order)
      } else {
        __builtin_amdgcn_s_wait_tensorcnt((short)0);
      }
    }
    __syncthreads();                                   // B tile visible to all waves

    const int k0 = s << 5;
    v16bf a = load_frag16(aP + k0, aP + k0 + 16);

    // B frags from LDS: column nn = 16j + l16, K chunks {8*hl..+7, 16+8*hl..+7}
    const u16* bp  = &bsm[s & 1][0];
    const u16* b0p = bp + (l16 << 5) + (hl << 3);
    const u16* b1p = b0p + (16 << 5);
    const u16* b2p = b0p + (32 << 5);
    const u16* b3p = b0p + (48 << 5);
    v16bf b0 = load_frag16(b0p, b0p + 16);
    v16bf b1 = load_frag16(b1p, b1p + 16);
    v16bf b2 = load_frag16(b2p, b2p + 16);
    v16bf b3 = load_frag16(b3p, b3p + 16);

    acc0 = __builtin_amdgcn_wmma_f32_16x16x32_bf16(false, a, false, b0, (short)0, acc0, false, false);
    acc1 = __builtin_amdgcn_wmma_f32_16x16x32_bf16(false, a, false, b1, (short)0, acc1, false, false);
    acc2 = __builtin_amdgcn_wmma_f32_16x16x32_bf16(false, a, false, b2, (short)0, acc2, false, false);
    acc3 = __builtin_amdgcn_wmma_f32_16x16x32_bf16(false, a, false, b3, (short)0, acc3, false, false);

    __syncthreads();                                   // done reading buf[s&1]
  }

  // C/D layout: VGPR r -> row mBase + r + 8*hl, col nBase + 16j + l16
  const int row0 = mBase + (hl << 3);
#pragma unroll
  for (int j = 0; j < 4; ++j) {
    v8f acc = (j == 0) ? acc0 : (j == 1) ? acc1 : (j == 2) ? acc2 : acc3;
    const int col = nBase + 16 * j + l16;
    const float bv = (bias != nullptr && col < nbias) ? bias[col] : 0.f;
#pragma unroll
    for (int r = 0; r < 8; ++r) {
      float y = acc[r] + bv;
      const size_t o = (size_t)(row0 + r) * ldo + col;
      if (act == 1) {
        y = (y > 0.f) ? y : 0.f;
        outB[o] = f2bf(y);
      } else {
        if (act == 2)      y = sigmoidf_(y);
        else if (act == 3) y = softplusf_(y);
        else if (act == 4) y = softplusf_(y) + 1.f;
        outF[o] = y;
      }
    }
  }
}

// ---------------- conversion / packing kernels ----------------
__global__ void conv_pad(const float* __restrict__ s, u16* __restrict__ d,
                         int Ndst, int Nsrc, int K, int Kpad) {
  int i = blockIdx.x * blockDim.x + threadIdx.x;
  int tot = Ndst * Kpad;
  if (i >= tot) return;
  int n = i / Kpad, k = i % Kpad;
  d[i] = (n < Nsrc && k < K) ? f2bf(s[(size_t)n * K + k]) : (u16)0;
}

// packed LSTM weight: K order [phi_x(512) | h(512) | phi_z(512)]
__global__ void conv_lstm(const float* __restrict__ wih, const float* __restrict__ whh,
                          u16* __restrict__ d) {
  int i = blockIdx.x * blockDim.x + threadIdx.x;
  if (i >= 2048 * 1536) return;
  int n = i / 1536, k = i % 1536;
  float v;
  if (k < 512)       v = wih[(size_t)n * 1024 + k];
  else if (k < 1024) v = whh[(size_t)n * 512 + (k - 512)];
  else               v = wih[(size_t)n * 1024 + 512 + (k - 1024)];
  d[i] = f2bf(v);
}

// dec_w1 with K halves swapped: input becomes [h | phi_z]
__global__ void conv_decw1(const float* __restrict__ s, u16* __restrict__ d) {
  int i = blockIdx.x * blockDim.x + threadIdx.x;
  if (i >= 512 * 1024) return;
  int n = i / 1024, k = i % 1024;
  float v = (k < 512) ? s[(size_t)n * 1024 + 512 + k] : s[(size_t)n * 1024 + (k - 512)];
  d[i] = f2bf(v);
}

// x_t [512,88] f32 -> [512,96] bf16 (zero-padded K)
__global__ void conv_x(const float* __restrict__ x, u16* __restrict__ d, int t) {
  int i = blockIdx.x * blockDim.x + threadIdx.x;
  if (i >= 512 * 96) return;
  int b = i / 96, k = i % 96;
  d[i] = (k < 88) ? f2bf(x[((size_t)t * 512 + b) * 88 + k]) : (u16)0;
}

__global__ void zero_f32(float* __restrict__ p, int n) {
  int i = blockIdx.x * blockDim.x + threadIdx.x;
  if (i < n) p[i] = 0.f;
}

__global__ void zero_cat_h(u16* __restrict__ cat) {
  int i = blockIdx.x * blockDim.x + threadIdx.x;
  if (i >= 512 * 512) return;
  int b = i >> 9, j = i & 511;
  cat[(size_t)b * 1536 + 512 + j] = 0;
}

// ---------------- elementwise kernels ----------------
__global__ __launch_bounds__(128) void e_zkld(
    const float* __restrict__ em, const float* __restrict__ es,
    const float* __restrict__ pm, const float* __restrict__ ps,
    const float* __restrict__ eps, const float* __restrict__ mask,
    u16* __restrict__ zbf, float* __restrict__ kld, int t)
{
  __shared__ float sm[128];
  int b = blockIdx.x, j = threadIdx.x;
  int idx = b * 128 + j;
  float m = em[idx], s = es[idx], qm = pm[idx], qs = ps[idx];
  float e = eps[((size_t)t * 512 + b) * 128 + j];
  zbf[idx] = f2bf(e * s + m);
  float dm = m - qm;
  float term = 2.f * (logf(qs) - logf(s)) + (s * s + dm * dm) / (qs * qs) - 1.f;
  sm[j] = term;
  __syncthreads();
  for (int st = 64; st > 0; st >>= 1) { if (j < st) sm[j] += sm[j + st]; __syncthreads(); }
  if (j == 0) kld[b] += 0.5f * sm[0] * mask[(size_t)t * 512 + b];
}

__global__ __launch_bounds__(128) void e_nll(
    const float* __restrict__ dm, const float* __restrict__ x,
    const float* __restrict__ mask, float* __restrict__ nll, int t)
{
  __shared__ float sm[128];
  int b = blockIdx.x, j = threadIdx.x;
  float term = 0.f;
  if (j < 88) {
    float p  = dm[b * 128 + j];
    float xv = x[((size_t)t * 512 + b) * 88 + j];
    term = xv * logf(p) + (1.f - xv) * logf(1.f - p);
  }
  sm[j] = term;
  __syncthreads();
  for (int st = 64; st > 0; st >>= 1) { if (j < st) sm[j] += sm[j + st]; __syncthreads(); }
  if (j == 0) nll[b] += -sm[0] * mask[(size_t)t * 512 + b];
}

__global__ void e_lstm(const float* __restrict__ g, float* __restrict__ c,
                       float* __restrict__ hf, u16* __restrict__ cat)
{
  int i = blockIdx.x * blockDim.x + threadIdx.x;
  if (i >= 512 * 512) return;
  int b = i >> 9, j = i & 511;
  const float* gr = g + (size_t)b * 2048;
  float ig = gr[j], fg = gr[512 + j], gg = gr[1024 + j], og = gr[1536 + j];
  float cn = sigmoidf_(fg) * c[i] + sigmoidf_(ig) * tanhf(gg);
  float hn = sigmoidf_(og) * tanhf(cn);
  c[i]  = cn;
  hf[i] = hn;
  cat[(size_t)b * 1536 + 512 + j] = f2bf(hn);
}

__global__ __launch_bounds__(512) void finalize_k(
    const float* __restrict__ kld, const float* __restrict__ nll, float* __restrict__ out)
{
  __shared__ float sm[512];
  int t = threadIdx.x;
  float kv = kld[t], nv = nll[t];
  out[1 + t] = -nv;
  out[513 + 262144 + t] = kv;
  sm[t] = kv + nv;
  __syncthreads();
  for (int s = 256; s > 0; s >>= 1) { if (t < s) sm[t] += sm[t + s]; __syncthreads(); }
  if (t == 0) out[0] = sm[0];
}

__global__ void copy_h(const float* __restrict__ hf, float* __restrict__ out) {
  int i = blockIdx.x * blockDim.x + threadIdx.x;
  if (i < 262144) out[513 + i] = hf[i];
}

// ---------------- host orchestration ----------------
static inline size_t alignUp(size_t x) { return (x + 255) & ~(size_t)255; }

static void launch_gemm(hipStream_t st, const u16* A, int lda, const u16* W, int ldw,
                        const float* bias, int nbias, float* outF, u16* outB, int ldo,
                        int K, int N, int act)
{
  dim3 grid(4 /* 512/128 */, N / 64), block(256);
  gemm_wmma_bf16<<<grid, block, 0, st>>>(A, lda, W, ldw, N, bias, nbias, outF, outB, ldo, K, act);
}

extern "C" void kernel_launch(void* const* d_in, const int* in_sizes, int n_in,
                              void* d_out, int out_size, void* d_ws, size_t ws_size,
                              hipStream_t stream)
{
  (void)in_sizes; (void)n_in; (void)out_size; (void)ws_size;
  const float* x    = (const float*)d_in[0];   // [128,512,88]
  const float* mask = (const float*)d_in[1];   // [128,512]
  const float* eps  = (const float*)d_in[2];   // [128,512,128]
  int pi = 4;                                  // d_in[3] = num_particles (unused)
  const float* phi_x_w1 = (const float*)d_in[pi++]; const float* phi_x_b1 = (const float*)d_in[pi++];
  const float* phi_x_w2 = (const float*)d_in[pi++]; const float* phi_x_b2 = (const float*)d_in[pi++];
  const float* phi_z_w  = (const float*)d_in[pi++]; const float* phi_z_b  = (const float*)d_in[pi++];
  const float* enc_w1   = (const float*)d_in[pi++]; const float* enc_b1   = (const float*)d_in[pi++];
  const float* enc_w2   = (const float*)d_in[pi++]; const float* enc_b2   = (const float*)d_in[pi++];
  const float* enc_mean_w = (const float*)d_in[pi++]; const float* enc_mean_b = (const float*)d_in[pi++];
  const float* enc_std_w  = (const float*)d_in[pi++]; const float* enc_std_b  = (const float*)d_in[pi++];
  const float* prior_w    = (const float*)d_in[pi++]; const float* prior_b    = (const float*)d_in[pi++];
  const float* prior_mean_w = (const float*)d_in[pi++]; const float* prior_mean_b = (const float*)d_in[pi++];
  const float* prior_std_w  = (const float*)d_in[pi++]; const float* prior_std_b  = (const float*)d_in[pi++];
  const float* dec_w1   = (const float*)d_in[pi++]; const float* dec_b1   = (const float*)d_in[pi++];
  const float* dec_w2   = (const float*)d_in[pi++]; const float* dec_b2   = (const float*)d_in[pi++];
  const float* dec_mean_w = (const float*)d_in[pi++]; const float* dec_mean_b = (const float*)d_in[pi++];
  pi += 2;                                     // dec_std_w/b unused by reference step
  const float* lstm_w_ih = (const float*)d_in[pi++];
  const float* lstm_w_hh = (const float*)d_in[pi++];

  float* out = (float*)d_out;

  // -------- workspace carve-up --------
  char* p = (char*)d_ws;
  auto takeU16 = [&](size_t elems) { u16* r = (u16*)p; p += alignUp(elems * 2); return r; };
  auto takeF32 = [&](size_t elems) { float* r = (float*)p; p += alignUp(elems * 4); return r; };

  u16* w_phi_x1 = takeU16(512 * 96);     // K padded 88->96
  u16* w_phi_x2 = takeU16(512 * 512);
  u16* w_phi_z  = takeU16(512 * 128);
  u16* w_enc1   = takeU16(512 * 1024);
  u16* w_enc2   = takeU16(512 * 512);
  u16* w_encm   = takeU16(128 * 512);
  u16* w_encs   = takeU16(128 * 512);
  u16* w_pri    = takeU16(512 * 512);
  u16* w_prim   = takeU16(128 * 512);
  u16* w_pris   = takeU16(128 * 512);
  u16* w_dec1   = takeU16(512 * 1024);   // K-swapped
  u16* w_dec2   = takeU16(512 * 512);
  u16* w_decm   = takeU16(128 * 512);    // N padded 88->128
  u16* w_lstm   = takeU16(2048 * 1536);  // packed [Wih_x | Whh | Wih_z]

  u16* cat  = takeU16(512 * 1536);       // [phi_x | h | phi_z] per row
  u16* t0   = takeU16(512 * 512);
  u16* t1   = takeU16(512 * 512);
  u16* zbf  = takeU16(512 * 128);
  u16* xpad = takeU16(512 * 96);

  float* encM = takeF32(512 * 128);
  float* encS = takeF32(512 * 128);
  float* priM = takeF32(512 * 128);
  float* priS = takeF32(512 * 128);
  float* decM = takeF32(512 * 128);
  float* gatesF = takeF32(512 * 2048);
  float* hF  = takeF32(512 * 512);
  float* cF  = takeF32(512 * 512);
  float* kld = takeF32(512);
  float* nll = takeF32(512);

  const int TB = 256;
  auto nb = [](int n) { return (n + 255) / 256; };

  // -------- weight conversion (bf16, pad/pack) --------
  conv_pad<<<nb(512 * 96), TB, 0, stream>>>(phi_x_w1, w_phi_x1, 512, 512, 88, 96);
  conv_pad<<<nb(512 * 512), TB, 0, stream>>>(phi_x_w2, w_phi_x2, 512, 512, 512, 512);
  conv_pad<<<nb(512 * 128), TB, 0, stream>>>(phi_z_w, w_phi_z, 512, 512, 128, 128);
  conv_pad<<<nb(512 * 1024), TB, 0, stream>>>(enc_w1, w_enc1, 512, 512, 1024, 1024);
  conv_pad<<<nb(512 * 512), TB, 0, stream>>>(enc_w2, w_enc2, 512, 512, 512, 512);
  conv_pad<<<nb(128 * 512), TB, 0, stream>>>(enc_mean_w, w_encm, 128, 128, 512, 512);
  conv_pad<<<nb(128 * 512), TB, 0, stream>>>(enc_std_w, w_encs, 128, 128, 512, 512);
  conv_pad<<<nb(512 * 512), TB, 0, stream>>>(prior_w, w_pri, 512, 512, 512, 512);
  conv_pad<<<nb(128 * 512), TB, 0, stream>>>(prior_mean_w, w_prim, 128, 128, 512, 512);
  conv_pad<<<nb(128 * 512), TB, 0, stream>>>(prior_std_w, w_pris, 128, 128, 512, 512);
  conv_decw1<<<nb(512 * 1024), TB, 0, stream>>>(dec_w1, w_dec1);
  conv_pad<<<nb(512 * 512), TB, 0, stream>>>(dec_w2, w_dec2, 512, 512, 512, 512);
  conv_pad<<<nb(128 * 512), TB, 0, stream>>>(dec_mean_w, w_decm, 128, 88, 512, 512);
  conv_lstm<<<nb(2048 * 1536), TB, 0, stream>>>(lstm_w_ih, lstm_w_hh, w_lstm);

  // -------- state init --------
  zero_f32<<<nb(512 * 512), TB, 0, stream>>>(hF, 512 * 512);
  zero_f32<<<nb(512 * 512), TB, 0, stream>>>(cF, 512 * 512);
  zero_f32<<<nb(512), TB, 0, stream>>>(kld, 512);
  zero_f32<<<nb(512), TB, 0, stream>>>(nll, 512);
  zero_cat_h<<<nb(512 * 512), TB, 0, stream>>>(cat);

  // -------- time scan --------
  for (int t = 0; t < 128; ++t) {
    conv_x<<<nb(512 * 96), TB, 0, stream>>>(x, xpad, t);
    // phi_x -> cat[:,0:512]
    launch_gemm(stream, xpad, 96, w_phi_x1, 96, phi_x_b1, 512, nullptr, t0, 512, 96, 512, 1);
    launch_gemm(stream, t0, 512, w_phi_x2, 512, phi_x_b2, 512, nullptr, cat, 1536, 512, 512, 1);
    // encoder
    launch_gemm(stream, cat, 1536, w_enc1, 1024, enc_b1, 512, nullptr, t0, 512, 1024, 512, 1);
    launch_gemm(stream, t0, 512, w_enc2, 512, enc_b2, 512, nullptr, t1, 512, 512, 512, 1);
    launch_gemm(stream, t1, 512, w_encm, 512, enc_mean_b, 128, encM, nullptr, 128, 512, 128, 0);
    launch_gemm(stream, t1, 512, w_encs, 512, enc_std_b, 128, encS, nullptr, 128, 512, 128, 3);
    // prior from h (cat cols 512..1023)
    launch_gemm(stream, cat + 512, 1536, w_pri, 512, prior_b, 512, nullptr, t0, 512, 512, 512, 1);
    launch_gemm(stream, t0, 512, w_prim, 512, prior_mean_b, 128, priM, nullptr, 128, 512, 128, 0);
    launch_gemm(stream, t0, 512, w_pris, 512, prior_std_b, 128, priS, nullptr, 128, 512, 128, 4);
    // z sample + KLD
    e_zkld<<<512, 128, 0, stream>>>(encM, encS, priM, priS, eps, mask, zbf, kld, t);
    // phi_z -> cat[:,1024:1536]
    launch_gemm(stream, zbf, 128, w_phi_z, 128, phi_z_b, 512, nullptr, cat + 1024, 1536, 128, 512, 1);
    // decoder: input [h, phi_z] = cat[:,512:1536] with K-swapped dec_w1
    launch_gemm(stream, cat + 512, 1536, w_dec1, 1024, dec_b1, 512, nullptr, t0, 512, 1024, 512, 1);
    launch_gemm(stream, t0, 512, w_dec2, 512, dec_b2, 512, nullptr, t1, 512, 512, 512, 1);
    launch_gemm(stream, t1, 512, w_decm, 512, dec_mean_b, 88, decM, nullptr, 128, 512, 128, 2);
    e_nll<<<512, 128, 0, stream>>>(decM, x, mask, nll, t);
    // LSTM gates: single fused GEMM over packed K=1536
    launch_gemm(stream, cat, 1536, w_lstm, 1536, nullptr, 0, gatesF, nullptr, 2048, 1536, 2048, 0);
    e_lstm<<<nb(512 * 512), TB, 0, stream>>>(gatesF, cF, hF, cat);
  }

  // -------- outputs: [total | -nll | h | kld] --------
  finalize_k<<<1, 512, 0, stream>>>(kld, nll, out);
  copy_h<<<nb(262144), TB, 0, stream>>>(hF, out);
}
